// GeometryAwareSelfAttentionBlock_67242007986298
// MI455X (gfx1250) — matrix operand order repaired and checked by reading, not compiled, not run on hardware
//
#include <hip/hip_runtime.h>
#include <math.h>

// ---------------------------------------------------------------------------
// CDNA5 / gfx1250, wave32. WMMA f32 = f16 x f16 + f32, 16x16x32.
// ---------------------------------------------------------------------------
typedef __attribute__((ext_vector_type(16))) _Float16 v16h;
typedef __attribute__((ext_vector_type(8)))  float    v8f;

#define BB   2
#define NN   4096
#define DD   384
#define HH   6
#define HDD  64
#define KNN  8
#define MTOT (BB * NN)          // 8192 rows
#define CIN  (3 + DD)           // 387 channels in points / out

// ===========================================================================
// Small helpers
// ===========================================================================
__device__ __forceinline__ void blk_reduce2_128(float& s, float& q, float* sh) {
  #pragma unroll
  for (int m = 16; m >= 1; m >>= 1) {
    s += __shfl_xor(s, m, 32);
    q += __shfl_xor(q, m, 32);
  }
  const int wv = threadIdx.x >> 5, ln = threadIdx.x & 31;
  if (ln == 0) { sh[wv] = s; sh[4 + wv] = q; }
  __syncthreads();
  s = sh[0] + sh[1] + sh[2] + sh[3];
  q = sh[4] + sh[5] + sh[6] + sh[7];
}

// Max-reduce across a 16-lane row using DPP swizzles (no LDS round trips).
// quad_perm xor1 / xor2, then row_half_mirror (pairs 8-lane halves), then
// row_mirror (pairs 16-lane halves). All stay within a 16-lane DPP row.
__device__ __forceinline__ float row16_max(float v) {
#if __has_builtin(__builtin_amdgcn_update_dpp)
  int p;
  p = __builtin_amdgcn_update_dpp(0, __float_as_int(v), 0xB1, 0xF, 0xF, true);
  v = fmaxf(v, __int_as_float(p));   // xor 1
  p = __builtin_amdgcn_update_dpp(0, __float_as_int(v), 0x4E, 0xF, 0xF, true);
  v = fmaxf(v, __int_as_float(p));   // xor 2
  p = __builtin_amdgcn_update_dpp(0, __float_as_int(v), 0x141, 0xF, 0xF, true);
  v = fmaxf(v, __int_as_float(p));   // row_half_mirror (within 8)
  p = __builtin_amdgcn_update_dpp(0, __float_as_int(v), 0x140, 0xF, 0xF, true);
  v = fmaxf(v, __int_as_float(p));   // row_mirror (within 16)
  return v;
#else
  #pragma unroll
  for (int msk = 1; msk < 16; msk <<= 1)
    v = fmaxf(v, __shfl_xor(v, msk, 32));
  return v;
#endif
}

// ===========================================================================
// Weight conversion kernels
// ===========================================================================
__global__ void cvt_f16_kernel(const float* __restrict__ src,
                               _Float16* __restrict__ dst, int n) {
  int i = blockIdx.x * 256 + threadIdx.x;
  if (i < n) dst[i] = (_Float16)src[i];
}

// knn_w is (768,384) row-major: W1 = rows 0..383, W2 = rows 384..767.
// Emit W1 (f16) and Wd = W2 - W1 (f16).
__global__ void knnw_kernel(const float* __restrict__ knn_w,
                            _Float16* __restrict__ w1,
                            _Float16* __restrict__ wd) {
  int i = blockIdx.x * 256 + threadIdx.x;
  if (i < DD * DD) {
    float a = knn_w[i];
    w1[i] = (_Float16)a;
    wd[i] = (_Float16)(knn_w[DD * DD + i] - a);
  }
}

__global__ void coords_copy_kernel(const float* __restrict__ pts,
                                   float* __restrict__ out) {
  int i = blockIdx.x * 256 + threadIdx.x;   // over B*3*N
  if (i < BB * 3 * NN) {
    int b = i / (3 * NN), r = i % (3 * NN);
    size_t off = (size_t)b * CIN * NN + r;
    out[off] = pts[off];
  }
}

// ===========================================================================
// LayerNorm 1: points[b, 3+c, n] (channel-major) -> nf16[b*n, c]
// ===========================================================================
__global__ __launch_bounds__(128)
void ln1_kernel(const float* __restrict__ pts, const float* __restrict__ g,
                const float* __restrict__ be, _Float16* __restrict__ nf16) {
  __shared__ float sh[8];
  const int p = blockIdx.x;
  const int b = p >> 12, n = p & (NN - 1);
  const float* base = pts + (size_t)b * CIN * NN + (size_t)3 * NN + n;
  float x[3];
  #pragma unroll
  for (int j = 0; j < 3; ++j)
    x[j] = base[(size_t)(threadIdx.x + j * 128) * NN];
  float s = x[0] + x[1] + x[2];
  float q = x[0] * x[0] + x[1] * x[1] + x[2] * x[2];
  blk_reduce2_128(s, q, sh);
  const float mean = s * (1.0f / DD);
  const float var  = q * (1.0f / DD) - mean * mean;
  const float rstd = rsqrtf(var + 1e-5f);
  #pragma unroll
  for (int j = 0; j < 3; ++j) {
    int c = threadIdx.x + j * 128;
    nf16[(size_t)p * DD + c] = (_Float16)((x[j] - mean) * rstd * g[c] + be[c]);
  }
}

// ===========================================================================
// Generic WMMA GEMM with double-buffered LDS staging.
// out = act(A[M,K] @ W[K,Nc] + bias) (+accum/resid)
// 256 threads = 8 waves; block tile 128 x 64; K-step 32.
// ===========================================================================
template <bool BIAS, bool GELU, bool ACCUM, bool OF16, bool OF32, bool RESID>
__global__ __launch_bounds__(256)
void gemm16_kernel(const _Float16* __restrict__ A, const _Float16* __restrict__ W,
                   const float* __restrict__ bias, _Float16* __restrict__ o16,
                   float* __restrict__ o32, const float* __restrict__ resid,
                   int M, int Nc, int Kd) {
  __shared__ _Float16 Asl[2][128][40];   // 80B row stride (16B aligned)
  __shared__ _Float16 Bsl[2][64][40];    // transposed: [n][k]
  const int t = threadIdx.x;
  const int lane = t & 31, wave = t >> 5;
  const int hlf = lane >> 4, r16 = lane & 15;
  const int m0 = blockIdx.y * 128;
  const int n0 = blockIdx.x * 64;

  auto stage = [&](int kb, int buf) {
    #pragma unroll
    for (int q = 0; q < 2; ++q) {
      int l = t + q * 256;
      int row = l >> 2, seg = l & 3;
      uint4 v = *(const uint4*)(A + (size_t)(m0 + row) * Kd + kb + seg * 8);
      *(uint4*)&Asl[buf][row][seg * 8] = v;
    }
    {
      int row = t >> 3, seg = t & 7;
      union { uint4 u; _Float16 h[8]; } tmp;
      tmp.u = *(const uint4*)(W + (size_t)(kb + row) * Nc + n0 + seg * 8);
      #pragma unroll
      for (int j = 0; j < 8; ++j) Bsl[buf][seg * 8 + j][row] = tmp.h[j];
    }
  };

  v8f acc[4];
  v8f zero = {};
  #pragma unroll
  for (int nb = 0; nb < 4; ++nb) acc[nb] = zero;

  const int nk = Kd >> 5;
  stage(0, 0);
  for (int ki = 0; ki < nk; ++ki) {
    __syncthreads();                       // staging of buf(ki) complete
    if (ki + 1 < nk) stage((ki + 1) * 32, (ki + 1) & 1);
    const int cur = ki & 1;

    v16h a;
    const int arow = wave * 16 + r16;
    #pragma unroll
    for (int j = 0; j < 8; ++j) {
      int kA = (j < 4) ? (8 * hlf + 2 * j) : (16 + 8 * hlf + 2 * (j - 4));
      a[2 * j]     = Asl[cur][arow][kA];
      a[2 * j + 1] = Asl[cur][arow][kA + 1];
    }
    #pragma unroll
    for (int nb = 0; nb < 4; ++nb) {
      v16h bf;
      const int col  = nb * 16 + r16;
      const int koff = 16 * hlf;
      #pragma unroll
      for (int tt = 0; tt < 16; ++tt) bf[tt] = Bsl[cur][col][(tt < 8) ? (koff / 2 + tt) : (16 + koff / 2 + tt - 8)];
      acc[nb] = __builtin_amdgcn_wmma_f32_16x16x32_f16(
          false, a, false, bf, (short)0, acc[nb], false, false);
    }
  }

  // Epilogue. C layout: element i -> row (i + 8*hlf), col (lane&15).
  #pragma unroll
  for (int nb = 0; nb < 4; ++nb) {
    const int gcol = n0 + nb * 16 + r16;
    const float bv = BIAS ? bias[gcol] : 0.0f;
    #pragma unroll
    for (int i = 0; i < 8; ++i) {
      const int grow = m0 + wave * 16 + i + 8 * hlf;
      const size_t off = (size_t)grow * Nc + gcol;
      float v = acc[nb][i] + bv;
      if (GELU)  v = 0.5f * v * (1.0f + erff(v * 0.70710678118654752f));
      if (ACCUM) v += o32[off];
      if (RESID) v += resid[off];
      if (OF32)  o32[off] = v;
      if (OF16)  o16[off] = (_Float16)v;
    }
  }
}

// ===========================================================================
// Flash attention: per (b, h, 64-query tile). 128 threads = 4 waves.
// qkv layout: [b*N + n][1152], q = +0, k = +384, v = +768, head h at h*64.
// Double-buffered K/V staging; softmax row-sum computed with WMMA (P @ ones);
// row-max via DPP swizzles; 1/sqrt(HD) folded into the Q fragments.
// ===========================================================================
__global__ __launch_bounds__(128)
void flash_attn_kernel(const _Float16* __restrict__ qkv,
                       _Float16* __restrict__ attnO) {
  __shared__ _Float16 Klds[2][64][72];    // [key][hd]
  __shared__ _Float16 Vlds[2][64][72];    // transposed: [hd][key]
  __shared__ _Float16 Plds[4][16][72];    // per-wave P tile [qrow][key]
  const int t = threadIdx.x, lane = t & 31, wave = t >> 5;
  const int hlf = lane >> 4, r16 = lane & 15;
  const int b = blockIdx.z, h = blockIdx.y, n0 = blockIdx.x * 64;

  // Q fragments (A layout), pre-scaled by 1/sqrt(HD)=0.125 (exact in f16)
  v16h aq[2];
  {
    const size_t qbase =
        ((size_t)(b * NN + n0 + wave * 16 + r16)) * 1152 + h * HDD;
    #pragma unroll
    for (int ks = 0; ks < 2; ++ks) {
      #pragma unroll
      for (int j = 0; j < 8; ++j) {
        int kA = ((j < 4) ? (8 * hlf + 2 * j) : (16 + 8 * hlf + 2 * (j - 4)))
                 + 32 * ks;
        aq[ks][2 * j]     = (_Float16)(0.125f * (float)qkv[qbase + kA]);
        aq[ks][2 * j + 1] = (_Float16)(0.125f * (float)qkv[qbase + kA + 1]);
      }
    }
  }
  // All-ones B fragment for WMMA row-sum of P
  v16h vone;
  #pragma unroll
  for (int tt = 0; tt < 16; ++tt) vone[tt] = (_Float16)1.0f;

  auto stage = [&](int kt, int buf) {
    #pragma unroll
    for (int q = 0; q < 4; ++q) {
      int l = t + q * 128;
      int row = l >> 3, seg = l & 7;
      const size_t gb =
          ((size_t)(b * NN + kt * 64 + row)) * 1152 + h * HDD + seg * 8;
      uint4 kv = *(const uint4*)(qkv + gb + 384);
      *(uint4*)&Klds[buf][row][seg * 8] = kv;
      union { uint4 u; _Float16 hh[8]; } tv;
      tv.u = *(const uint4*)(qkv + gb + 768);
      #pragma unroll
      for (int j = 0; j < 8; ++j) Vlds[buf][seg * 8 + j][row] = tv.hh[j];
    }
  };

  float m_run[8], l_run[8];
  v8f o[4];
  v8f zero = {};
  #pragma unroll
  for (int i = 0; i < 8; ++i) { m_run[i] = -3.0e38f; l_run[i] = 0.0f; }
  #pragma unroll
  for (int hb = 0; hb < 4; ++hb) o[hb] = zero;

  const int NT = NN / 64;
  stage(0, 0);
  for (int kt = 0; kt < NT; ++kt) {
    __syncthreads();                        // staging of buf(kt) complete
    const int cur = kt & 1;
    if (kt + 1 < NT) stage(kt + 1, cur ^ 1);
    if (kt + 2 < NT && t < 64) {            // prefetch 2 tiles ahead
      const size_t gp = ((size_t)(b * NN + (kt + 2) * 64 + t)) * 1152 + h * HDD;
      __builtin_prefetch(qkv + gp + 384, 0, 1);
      __builtin_prefetch(qkv + gp + 768, 0, 1);
    }

    // S = Q @ K^T  (B matrix: B[k=hd][n=key] = Klds[key][hd])
    v8f s[4];
    #pragma unroll
    for (int nb = 0; nb < 4; ++nb) {
      s[nb] = zero;
      #pragma unroll
      for (int ks = 0; ks < 2; ++ks) {
        v16h bf;
        #pragma unroll
        for (int tt = 0; tt < 16; ++tt)
          bf[tt] = Klds[cur][nb * 16 + r16][32 * ks + ((tt < 8) ? (8 * hlf + tt) : (16 + 8 * hlf + tt - 8))];
        s[nb] = __builtin_amdgcn_wmma_f32_16x16x32_f16(
            false, aq[ks], false, bf, (short)0, s[nb], false, false);
      }
    }

    // Online softmax. Row M = i + 8*hlf lives on the 16 lanes of this half.
    #pragma unroll
    for (int i = 0; i < 8; ++i) {
      float v = -3.0e38f;
      #pragma unroll
      for (int nb = 0; nb < 4; ++nb) v = fmaxf(v, s[nb][i]);
      v = row16_max(v);
      const float mn   = fmaxf(m_run[i], v);
      const float corr = __expf(m_run[i] - mn);
      m_run[i] = mn;
      l_run[i] *= corr;
      #pragma unroll
      for (int hb = 0; hb < 4; ++hb) o[hb][i] *= corr;
      #pragma unroll
      for (int nb = 0; nb < 4; ++nb)
        Plds[wave][i + 8 * hlf][nb * 16 + r16] =
            (_Float16)__expf(s[nb][i] - mn);
    }
    // Intra-wave P store -> load: DS ops are in-order per wave; no barrier.

    // O += P @ V, and row-sum of P via P @ ones (reuses ap fragments)
    v8f lsum = zero;
    #pragma unroll
    for (int ks = 0; ks < 2; ++ks) {
      v16h ap;
      #pragma unroll
      for (int j = 0; j < 8; ++j) {
        int kA = ((j < 4) ? (8 * hlf + 2 * j) : (16 + 8 * hlf + 2 * (j - 4)))
                 + 32 * ks;
        ap[2 * j]     = Plds[wave][r16][kA];
        ap[2 * j + 1] = Plds[wave][r16][kA + 1];
      }
      #pragma unroll
      for (int hb = 0; hb < 4; ++hb) {
        v16h bf;
        #pragma unroll
        for (int tt = 0; tt < 16; ++tt)
          bf[tt] = Vlds[cur][hb * 16 + r16][32 * ks + ((tt < 8) ? (8 * hlf + tt) : (16 + 8 * hlf + tt - 8))];
        o[hb] = __builtin_amdgcn_wmma_f32_16x16x32_f16(
            false, ap, false, bf, (short)0, o[hb], false, false);
      }
      lsum = __builtin_amdgcn_wmma_f32_16x16x32_f16(
          false, ap, false, vone, (short)0, lsum, false, false);
    }
    #pragma unroll
    for (int i = 0; i < 8; ++i) l_run[i] += lsum[i];
  }

  // Write normalized output: attnO[b*N + qrow][h*64 + hd]
  #pragma unroll
  for (int hb = 0; hb < 4; ++hb) {
    #pragma unroll
    for (int i = 0; i < 8; ++i) {
      const int qrow = n0 + wave * 16 + i + 8 * hlf;
      attnO[((size_t)(b * NN + qrow)) * DD + h * HDD + hb * 16 + r16] =
          (_Float16)(o[hb][i] / l_run[i]);
    }
  }
}

// ===========================================================================
// kNN top-8 by squared distance. One block (256 thr) per query point.
// ===========================================================================
__global__ __launch_bounds__(256)
void knn_topk_kernel(const float* __restrict__ pts, int* __restrict__ idxout) {
  const int p = blockIdx.x;
  const int b = p >> 12, n = p & (NN - 1);
  const int t = threadIdx.x;
  const float* base = pts + (size_t)b * CIN * NN;
  const float qx = base[n], qy = base[NN + n], qz = base[2 * NN + n];

  float bd[8]; int bi[8];
  #pragma unroll
  for (int j = 0; j < 8; ++j) { bd[j] = 3.0e38f; bi[j] = 0; }

  for (int m = t; m < NN; m += 256) {
    const float dx = base[m] - qx;
    const float dy = base[NN + m] - qy;
    const float dz = base[2 * NN + m] - qz;
    const float d = dx * dx + dy * dy + dz * dz;
    if (d < bd[7]) {
      bd[7] = d; bi[7] = m;
      #pragma unroll
      for (int j = 7; j >= 1; --j) {
        if (bd[j] < bd[j - 1]) {
          float tv = bd[j]; bd[j] = bd[j - 1]; bd[j - 1] = tv;
          int ti = bi[j]; bi[j] = bi[j - 1]; bi[j - 1] = ti;
        }
      }
    }
  }

  __shared__ float sd[256 * 8];
  __shared__ int   si[256 * 8];
  __shared__ float rv[256];
  __shared__ int   rt[256];
  __shared__ int   res[8];
  #pragma unroll
  for (int j = 0; j < 8; ++j) { sd[t * 8 + j] = bd[j]; si[t * 8 + j] = bi[j]; }
  int pos = 0;
  __syncthreads();

  for (int r = 0; r < 8; ++r) {
    rv[t] = (pos < 8) ? sd[t * 8 + pos] : 3.0e38f;
    rt[t] = t;
    __syncthreads();
    for (int s = 128; s > 0; s >>= 1) {
      if (t < s && rv[t + s] < rv[t]) { rv[t] = rv[t + s]; rt[t] = rt[t + s]; }
      __syncthreads();
    }
    const int w = rt[0];
    __syncthreads();
    if (t == w) { res[r] = si[t * 8 + pos]; ++pos; }
    __syncthreads();
  }
  if (t < 8) idxout[(size_t)p * KNN + t] = res[t];
}

// ===========================================================================
// Gather neighbors: geom[p,c] = max_k leakyrelu(Y1[nbr_k,c] + Yc[p,c])
// ===========================================================================
__global__ __launch_bounds__(128)
void geom_kernel(const _Float16* __restrict__ Y1, const _Float16* __restrict__ Yc,
                 const int* __restrict__ idx, _Float16* __restrict__ geom) {
  const int p = blockIdx.x;
  const int b = p >> 12;
  __shared__ int ii[8];
  if (threadIdx.x < 8) ii[threadIdx.x] = idx[(size_t)p * KNN + threadIdx.x];
  __syncthreads();
  for (int c = threadIdx.x; c < DD; c += 128) {
    const float yc = (float)Yc[(size_t)p * DD + c];
    float vm = -3.0e38f;
    #pragma unroll
    for (int k = 0; k < KNN; ++k) {
      const size_t row = (size_t)b * NN + ii[k];
      float v = (float)Y1[row * DD + c] + yc;
      v = (v > 0.0f) ? v : 0.2f * v;
      vm = fmaxf(vm, v);
    }
    geom[(size_t)p * DD + c] = (_Float16)vm;
  }
}

// ===========================================================================
// LN2 epilogue: x = merged + attnP; LN(x) -> f32 + f16
// ===========================================================================
__global__ __launch_bounds__(128)
void ln2_kernel(const float* __restrict__ merged, const float* __restrict__ attnP,
                const float* __restrict__ g, const float* __restrict__ be,
                float* __restrict__ out32, _Float16* __restrict__ out16) {
  __shared__ float sh[8];
  const int p = blockIdx.x;
  float x[3];
  #pragma unroll
  for (int j = 0; j < 3; ++j) {
    int c = threadIdx.x + j * 128;
    x[j] = merged[(size_t)p * DD + c] + attnP[(size_t)p * DD + c];
  }
  float s = x[0] + x[1] + x[2];
  float q = x[0] * x[0] + x[1] * x[1] + x[2] * x[2];
  blk_reduce2_128(s, q, sh);
  const float mean = s * (1.0f / DD);
  const float var  = q * (1.0f / DD) - mean * mean;
  const float rstd = rsqrtf(var + 1e-5f);
  #pragma unroll
  for (int j = 0; j < 3; ++j) {
    int c = threadIdx.x + j * 128;
    float v = (x[j] - mean) * rstd * g[c] + be[c];
    out32[(size_t)p * DD + c] = v;
    out16[(size_t)p * DD + c] = (_Float16)v;
  }
}

// ===========================================================================
// Transpose final [B,N,D] f32 -> out[b, 3+c, n]
// ===========================================================================
__global__ void transpose_out_kernel(const float* __restrict__ fin,
                                     float* __restrict__ out) {
  __shared__ float tile[32][33];
  const int b = blockIdx.z;
  const int n0 = blockIdx.x * 32, c0 = blockIdx.y * 32;
  const int tx = threadIdx.x, ty = threadIdx.y;   // (32, 8)
  #pragma unroll
  for (int i = 0; i < 4; ++i)
    tile[ty + i * 8][tx] =
        fin[((size_t)b * NN + n0 + ty + i * 8) * DD + c0 + tx];
  __syncthreads();
  #pragma unroll
  for (int i = 0; i < 4; ++i)
    out[(size_t)b * CIN * NN + (size_t)(3 + c0 + ty + i * 8) * NN + n0 + tx] =
        tile[tx][ty + i * 8];
}

// ===========================================================================
// Launch
// ===========================================================================
extern "C" void kernel_launch(void* const* d_in, const int* in_sizes, int n_in,
                              void* d_out, int out_size, void* d_ws, size_t ws_size,
                              hipStream_t stream) {
  const float* pts       = (const float*)d_in[0];
  const float* ln1_g     = (const float*)d_in[1];
  const float* ln1_b     = (const float*)d_in[2];
  const float* qkv_w     = (const float*)d_in[3];
  const float* attnout_w = (const float*)d_in[4];
  const float* attnout_b = (const float*)d_in[5];
  const float* knn_w     = (const float*)d_in[6];
  const float* knn_b     = (const float*)d_in[7];
  const float* merge_w   = (const float*)d_in[8];
  const float* merge_b   = (const float*)d_in[9];
  const float* ln2_g     = (const float*)d_in[10];
  const float* ln2_b     = (const float*)d_in[11];
  const float* ff1_w     = (const float*)d_in[12];
  const float* ff1_b     = (const float*)d_in[13];
  const float* ff2_w     = (const float*)d_in[14];
  const float* ff2_b     = (const float*)d_in[15];
  float* out = (float*)d_out;

  char* w = (char*)d_ws;
  // Workspace layout (bytes); aliased regions noted.
  _Float16* nf16    = (_Float16*)(w + 0);                 //  6.29 MB
  _Float16* qkv16   = (_Float16*)(w + 6291456);           // 18.87 MB (reused)
  _Float16* Y1_16   = (_Float16*)(w + 6291456);           //   alias qkv16+0
  _Float16* Yc16    = (_Float16*)(w + 6291456 + 6291456); //   alias qkv16+6.29M
  _Float16* geom16  = (_Float16*)(w + 6291456 + 12582912);//   alias qkv16+12.58M
  _Float16* attnO16 = (_Float16*)(w + 25165824);          //  6.29 MB (reused ln2_16)
  _Float16* ln2_16  = (_Float16*)(w + 25165824);
  float*    attnP32 = (float*)(w + 31457280);             // 12.58 MB
  _Float16* attnP16 = (_Float16*)(w + 44040192);          //  6.29 MB
  float*    merged32= (float*)(w + 50331648);             // 12.58 MB (reused final32)
  float*    final32 = (float*)(w + 50331648);
  float*    ln2_32  = (float*)(w + 62914560);             // 12.58 MB
  _Float16* ffh16   = (_Float16*)(w + 75497472);          // 12.58 MB
  int*      knnidx  = (int*)(w + 88080384);               //  0.26 MB
  char*     wt      = w + 88342528;                       //  ~3.5 MB f16 weights
  _Float16* qkvw16  = (_Float16*)(wt + 0);
  _Float16* aow16   = (_Float16*)(wt + 884736);
  _Float16* w1_16   = (_Float16*)(wt + 1179648);
  _Float16* wd16    = (_Float16*)(wt + 1474560);
  _Float16* wm1_16  = (_Float16*)(wt + 1769472);
  _Float16* wm2_16  = (_Float16*)(wt + 2064384);
  _Float16* ff1w16  = (_Float16*)(wt + 2359296);
  _Float16* ff2w16  = (_Float16*)(wt + 2949120);

  // --- weight conversion ---
  auto cvt = [&](const float* s, _Float16* d, int n) {
    cvt_f16_kernel<<<(n + 255) / 256, 256, 0, stream>>>(s, d, n);
  };
  cvt(qkv_w, qkvw16, DD * 3 * DD);
  cvt(attnout_w, aow16, DD * DD);
  cvt(merge_w, wm1_16, DD * DD);                 // rows 0..383
  cvt(merge_w + DD * DD, wm2_16, DD * DD);       // rows 384..767
  cvt(ff1_w, ff1w16, DD * 2 * DD);
  cvt(ff2_w, ff2w16, 2 * DD * DD);
  knnw_kernel<<<(DD * DD + 255) / 256, 256, 0, stream>>>(knn_w, w1_16, wd16);
  coords_copy_kernel<<<(BB * 3 * NN + 255) / 256, 256, 0, stream>>>(pts, out);

  // --- LN1 ---
  ln1_kernel<<<MTOT, 128, 0, stream>>>(pts, ln1_g, ln1_b, nf16);

  // --- qkv = nf @ qkv_w ---
  gemm16_kernel<false, false, false, true, false, false>
      <<<dim3(3 * DD / 64, MTOT / 128), 256, 0, stream>>>(
          nf16, qkvw16, nullptr, qkv16, nullptr, nullptr, MTOT, 3 * DD, DD);

  // --- flash attention ---
  flash_attn_kernel<<<dim3(NN / 64, HH, BB), 128, 0, stream>>>(qkv16, attnO16);

  // --- attn projection (f32 + f16 out) ---
  gemm16_kernel<true, false, false, true, true, false>
      <<<dim3(DD / 64, MTOT / 128), 256, 0, stream>>>(
          attnO16, aow16, attnout_b, attnP16, attnP32, nullptr, MTOT, DD, DD);

  // --- kNN branch: Y1 = nf@W1, Yc = nf@(W2-W1)+knn_b, topk, gather/max ---
  knn_topk_kernel<<<MTOT, 256, 0, stream>>>(pts, knnidx);
  gemm16_kernel<false, false, false, true, false, false>
      <<<dim3(DD / 64, MTOT / 128), 256, 0, stream>>>(
          nf16, w1_16, nullptr, Y1_16, nullptr, nullptr, MTOT, DD, DD);
  gemm16_kernel<true, false, false, true, false, false>
      <<<dim3(DD / 64, MTOT / 128), 256, 0, stream>>>(
          nf16, wd16, knn_b, Yc16, nullptr, nullptr, MTOT, DD, DD);
  geom_kernel<<<MTOT, 128, 0, stream>>>(Y1_16, Yc16, knnidx, geom16);

  // --- merged = attn@Wm1 + geom@Wm2 + merge_b ---
  gemm16_kernel<true, false, false, false, true, false>
      <<<dim3(DD / 64, MTOT / 128), 256, 0, stream>>>(
          attnP16, wm1_16, merge_b, nullptr, merged32, nullptr, MTOT, DD, DD);
  gemm16_kernel<false, false, true, false, true, false>
      <<<dim3(DD / 64, MTOT / 128), 256, 0, stream>>>(
          geom16, wm2_16, nullptr, nullptr, merged32, nullptr, MTOT, DD, DD);

  // --- LN2 over (merged + attnP) ---
  ln2_kernel<<<MTOT, 128, 0, stream>>>(merged32, attnP32, ln2_g, ln2_b,
                                       ln2_32, ln2_16);

  // --- FF: gelu(ln2@ff1+b)@ff2+b + ln2 ---
  gemm16_kernel<true, true, false, true, false, false>
      <<<dim3(2 * DD / 64, MTOT / 128), 256, 0, stream>>>(
          ln2_16, ff1w16, ff1_b, ffh16, nullptr, nullptr, MTOT, 2 * DD, DD);
  gemm16_kernel<true, false, false, false, true, true>
      <<<dim3(DD / 64, MTOT / 128), 256, 0, stream>>>(
          ffh16, ff2w16, ff2_b, nullptr, final32, ln2_32, MTOT, DD, 2 * DD);

  // --- transpose to output layout ---
  transpose_out_kernel<<<dim3(NN / 32, DD / 32, BB), dim3(32, 8), 0, stream>>>(
      final32, out);
}
